// StabilizedSystemDoublePendulum_50096498541098
// MI455X (gfx1250) — matrix operand order repaired and checked by reading, not compiled, not run on hardware
//
#include <hip/hip_runtime.h>

typedef __attribute__((ext_vector_type(16))) _Float16 v16h;
typedef __attribute__((ext_vector_type(8)))  _Float16 v8h;
typedef __attribute__((ext_vector_type(8)))  float    v8f;

#define HID 64

__device__ __forceinline__ float sqf(float a) { return a * a; }

// Hardware tanh on gfx1250 (V_TANH_F32) if the builtin exists in this toolchain.
__device__ __forceinline__ float fast_tanh(float x) {
#if __has_builtin(__builtin_amdgcn_tanhf)
  return __builtin_amdgcn_tanhf(x);
#elif __has_builtin(__builtin_amdgcn_tanh_f32)
  return __builtin_amdgcn_tanh_f32(x);
#else
  return tanhf(x);
#endif
}

constexpr float cL1 = 1.0f, cL2 = 1.0f, cM1 = 1.0f, cM2 = 1.0f;
constexpr float cG = 9.81f, cK1 = 0.5f, cAE = 1.0f, cBD = 0.0f;
constexpr float cPI = 3.14159265358979323846f;

// ---------------- prep kernel: stage f16 weights + compute E_des ----------------
__global__ __launch_bounds__(256) void dp_prep(
    const float* __restrict__ W1, const float* __restrict__ b1,
    const float* __restrict__ W2, const float* __restrict__ b2,
    const float* __restrict__ W3, const float* __restrict__ b3,
    const float* __restrict__ x0,
    _Float16* __restrict__ w2fT,   // [n][k] = W2[k][n]  (forward B source)
    _Float16* __restrict__ w2b,    // [n][k] = W2[n][k]  (backward B source, row-major copy)
    float* __restrict__ eDes) {
  int tid = threadIdx.x;
  for (int i = tid; i < HID * HID; i += 256) {
    int n = i >> 6, k = i & 63;
    w2fT[i] = (_Float16)W2[k * HID + n];
    w2b[i]  = (_Float16)W2[i];
  }
  __shared__ float red[HID];
  if (tid < HID) {
    float q01 = x0[0], q02 = x0[1];
    float acc = b2[tid];
    for (int k = 0; k < HID; ++k) {
      float h1k = fast_tanh(fmaf(q02, W1[HID + k], fmaf(q01, W1[k], b1[k])));
      acc = fmaf(h1k, W2[k * HID + tid], acc);
    }
    red[tid] = fast_tanh(acc) * W3[tid];
  }
  __syncthreads();
  if (tid == 0) {
    float V0 = b3[0];
    for (int j = 0; j < HID; ++j) V0 += red[j];
    float q01 = x0[0], q02 = x0[1], p01 = x0[2], p02 = x0[3];
    float c2 = __cosf(q02);
    float pot0 = -cG * (__cosf(q01 + q02) * cL2 * cM2 + __cosf(q01) * cL1 * (cM1 + cM2))
               + cK1 * sqf(0.5f * cPI - q02);
    float M11 = 2.0f * c2 * cL1 * cL2 * cM2 + cL2 * cL2 * cM2 + cL1 * cL1 * (cM1 + cM2);
    float M12 = cL2 * (c2 * cL1 + cL2) * cM2;
    float M22 = cL2 * cL2 * cM2;
    float DEN = M11 * M22 - M12 * M12;
    float i11 = M22 / DEN, i12 = -M12 / DEN, i22 = M11 / DEN;
    float kin0 = 0.5f * (i11 * p01 * p01 + 2.0f * i12 * p01 * p02 + i22 * p02 * p02);
    eDes[0] = pot0 + kin0 + V0;
  }
}

// ---------------- fragment loaders (contiguous k order, same for A and B) ----------------
// A-matrix 16x32 f16: lane<16 holds M=lane, element e -> K = 32*kk + 16*(lane>>4) + e  (contiguous span)
__device__ __forceinline__ v16h load_afrag(const _Float16* rowp, int kk, int hi) {
  const v8h lo = *(const v8h*)(rowp + 32 * kk + 16 * hi);        // K = 32kk+16hi+0..7
  const v8h hh = *(const v8h*)(rowp + 32 * kk + 16 * hi + 8);   // K = 32kk+16hi+8..15
  v16h a;
#pragma unroll
  for (int i = 0; i < 8; ++i) { a[i] = lo[i]; a[i + 8] = hh[i]; }
  return a;
}
// B-matrix 32x16 f16: lane&15 = N, element e -> K = 16*(lane>>4) + e  (contiguous span)
__device__ __forceinline__ v16h load_bfrag(const _Float16* p) {
  const v8h lo = *(const v8h*)p;
  const v8h hh = *(const v8h*)(p + 8);
  v16h b;
#pragma unroll
  for (int i = 0; i < 8; ++i) { b[i] = lo[i]; b[i + 8] = hh[i]; }
  return b;
}

// ---------------- main kernel ----------------
__global__ __launch_bounds__(256) void dp_main(
    const float* __restrict__ x,
    const float* __restrict__ W1, const float* __restrict__ b1,
    const float* __restrict__ b2, const float* __restrict__ W3,
    const float* __restrict__ b3, const float* __restrict__ Tp,
    const _Float16* __restrict__ w2fT, const _Float16* __restrict__ w2b,
    const float* __restrict__ eDes,
    float* __restrict__ out, int N) {
  __shared__ _Float16 sW2fT[HID * HID];
  __shared__ _Float16 sW2b[HID * HID];
  __shared__ float sW1a[HID], sW1b[HID], sB1[HID], sB2[HID], sW3[HID];
  __shared__ _Float16 sH1[8][16 * HID];
  __shared__ _Float16 sD2[8][16 * HID];
  __shared__ float sRow[8][16][4];

  int tid = threadIdx.x;
  for (int i = tid; i < HID * HID; i += 256) { sW2fT[i] = w2fT[i]; sW2b[i] = w2b[i]; }
  if (tid < HID) {
    sW1a[tid] = W1[tid]; sW1b[tid] = W1[HID + tid];
    sB1[tid] = b1[tid];  sB2[tid] = b2[tid];  sW3[tid] = W3[tid];
  }
  __syncthreads();

  int wave = tid >> 5, lane = tid & 31;
  int nTiles = (N + 15) >> 4;
  int tile = blockIdx.x * 8 + wave;
  if (tile >= nTiles) return;

  int r = lane & 15, hi = lane >> 4;
  int base = tile * 16;
  int row = base + r;
  int rowC = min(row, N - 1);
  float4 xv = ((const float4*)x)[rowC];
  float q1 = xv.x, q2 = xv.y, p1 = xv.z, p2 = xv.w;

  _Float16* h1w = sH1[wave];
  _Float16* d2w = sD2[wave];

  // ---- layer 1: h1 = tanh(q@W1 + b1); lane covers cols [32*hi, 32*hi+32) of its row
  {
    int cbase = 32 * hi;
    _Float16 tmp[32];
#pragma unroll
    for (int i = 0; i < 32; ++i) {
      int c = cbase + i;
      tmp[i] = (_Float16)fast_tanh(fmaf(q2, sW1b[c], fmaf(q1, sW1a[c], sB1[c])));
    }
#pragma unroll
    for (int j = 0; j < 4; ++j) {
      v8h v;
#pragma unroll
      for (int e = 0; e < 8; ++e) v[e] = tmp[8 * j + e];
      *(v8h*)(h1w + r * HID + cbase + 8 * j) = v;
    }
  }
  asm volatile("s_wait_dscnt 0x0" ::: "memory");

  // ---- layer 2 forward GEMM:  pre2 = h1 @ W2   (16x64 = [16x64]x[64x64])
  v16h aF0 = load_afrag(h1w + r * HID, 0, hi);
  v16h aF1 = load_afrag(h1w + r * HID, 1, hi);
  const v8f zero = {0.f, 0.f, 0.f, 0.f, 0.f, 0.f, 0.f, 0.f};
  float vpart[8];
#pragma unroll
  for (int v = 0; v < 8; ++v) vpart[v] = 0.f;

#pragma unroll
  for (int c = 0; c < 4; ++c) {
    int n = 16 * c + r;
    const _Float16* bp = sW2fT + n * HID + 16 * hi;
    v16h b0 = load_bfrag(bp);
    v16h b1f = load_bfrag(bp + 32);
    v8f acc = __builtin_amdgcn_wmma_f32_16x16x32_f16(false, aF0, false, b0, (short)0, zero, false, false);
    acc = __builtin_amdgcn_wmma_f32_16x16x32_f16(false, aF1, false, b1f, (short)0, acc, false, false);
    float w3n = sW3[n], b2n = sB2[n];
#pragma unroll
    for (int v = 0; v < 8; ++v) {
      float h2 = fast_tanh(acc[v] + b2n);
      vpart[v] = fmaf(h2, w3n, vpart[v]);
      float d2 = w3n * (1.0f - h2 * h2);           // dV/d(pre2)
      d2w[(v + 8 * hi) * HID + n] = (_Float16)d2;
    }
  }
  asm volatile("s_wait_dscnt 0x0" ::: "memory");

  // ---- backward GEMM: g1 = d2 @ W2^T
  v16h aB0 = load_afrag(d2w + r * HID, 0, hi);
  v16h aB1 = load_afrag(d2w + r * HID, 1, hi);
  float p1part[8], p2part[8];
#pragma unroll
  for (int v = 0; v < 8; ++v) { p1part[v] = 0.f; p2part[v] = 0.f; }

#pragma unroll
  for (int c = 0; c < 4; ++c) {
    int n = 16 * c + r;                      // h1 index (output column)
    const _Float16* bp = sW2b + n * HID + 16 * hi;
    v16h b0 = load_bfrag(bp);
    v16h b1b = load_bfrag(bp + 32);
    v8f g = __builtin_amdgcn_wmma_f32_16x16x32_f16(false, aB0, false, b0, (short)0, zero, false, false);
    g = __builtin_amdgcn_wmma_f32_16x16x32_f16(false, aB1, false, b1b, (short)0, g, false, false);
    float w1an = sW1a[n], w1bn = sW1b[n];
#pragma unroll
    for (int v = 0; v < 8; ++v) {
      float hv = (float)h1w[(v + 8 * hi) * HID + n];
      float d1 = g[v] * (1.0f - hv * hv);
      p1part[v] = fmaf(d1, w1an, p1part[v]);
      p2part[v] = fmaf(d1, w1bn, p2part[v]);
    }
  }

  // ---- reduce V, dVdq1, dVdq2 across the 16 lanes of each half
#pragma unroll
  for (int v = 0; v < 8; ++v) {
    float a = vpart[v], bb = p1part[v], cc = p2part[v];
#pragma unroll
    for (int m = 1; m < 16; m <<= 1) {
      a += __shfl_xor(a, m, 16);
      bb += __shfl_xor(bb, m, 16);
      cc += __shfl_xor(cc, m, 16);
    }
    if (r == 0) {
      sRow[wave][v + 8 * hi][0] = a;
      sRow[wave][v + 8 * hi][1] = bb;
      sRow[wave][v + 8 * hi][2] = cc;
    }
  }
  asm volatile("s_wait_dscnt 0x0" ::: "memory");

  // ---- per-row physics (lanes 0..15 only; WMMA region is over, divergence OK)
  if (hi == 0 && row < N) {
    float V   = sRow[wave][r][0] + b3[0];
    float dV1 = sRow[wave][r][1];
    float dV2 = sRow[wave][r][2];

    float cq2 = __cosf(q2), sq2v = __sinf(q2);
    float M11 = 2.0f * cq2 * cL1 * cL2 * cM2 + cL2 * cL2 * cM2 + cL1 * cL1 * (cM1 + cM2);
    float M12 = cL2 * (cq2 * cL1 + cL2) * cM2;
    float M22 = cL2 * cL2 * cM2;
    float DEN = M11 * M22 - M12 * M12;
    float i11 = M22 / DEN, i12 = -M12 / DEN, i22 = M11 / DEN;
    float quad = i11 * p1 * p1 + 2.0f * i12 * p1 * p2 + i22 * p2 * p2;
    float kin = 0.5f * quad;
    float pot = -cG * (__cosf(q1 + q2) * cL2 * cM2 + __cosf(q1) * cL1 * (cM1 + cM2))
              + cK1 * sqf(0.5f * cPI - q2);
    float E = pot + kin + V;

    float pn = sqrtf(quad);
    float safe = (pn > 0.f) ? pn : 1.0f;
    float invs = 1.0f / safe;
    float ph1 = p1 * invs, ph2 = p2 * invs;

    float es = cAE * (eDes[0] - E);
    float u1 = -dV1 + es * ph1;
    float u2 = -dV2 + es * ph2;

    float DET = cL1 * cL1 * cL2 * cL2 * cM2 * cM2 * (-cq2 * cq2)
              + cL1 * cL1 * cL2 * cL2 * cM2 * cM2
              + cL1 * cL1 * cL2 * cL2 * cM1 * cM2;
    float dq1dt = (cL2 * cL2 * cM2 * p1 - cL2 * cM2 * p2 * (cL1 * cq2 + cL2)) / DET;
    float dq2dt = (p2 * (2.0f * cL2 * cL1 * cM2 * cq2 + cL1 * cL1 * (cM1 + cM2) + cL2 * cL2 * cM2)
                 - cL2 * cM2 * p1 * (cL1 * cq2 + cL2)) / DET;
    float dp1dt = -cG * (__sinf(q1 + q2) * cL2 * cM2 + __sinf(q1) * cL1 * (cM1 + cM2))
                + u1 - cBD * dq1dt;
    float t = cM1 + sq2v * sq2v * cM2;
    float denom = cL1 * cL1 * cL2 * cL2 * t * t;
    float inner = -p2 * p2 * sq2v * (cq2 * cL1 + cL2) * (cq2 * cL2 * cM2 + cL1 * (cM1 + cM2))
                + 0.5f * p1 * p2 * sq2v * cL2 *
                  (4.0f * cq2 * cL2 * cM2 + cL1 * (2.0f * cM1 + (3.0f + __cosf(2.0f * q2)) * cM2))
                + cL2 * cL2 * (-cq2 * p1 * p1 * sq2v * cM2
                               + cL1 * cL1 * t * t * (cG * __sinf(q1 + q2) * cL2 * cM2
                                                      - cK1 * (cPI - 2.0f * q2)));
    float dp2dt = -(inner / denom) + u2 - cBD * dq2dt;

    float Ta = fabsf(Tp[0]);
    float4 o;
    o.x = Ta * dq1dt; o.y = Ta * dq2dt; o.z = Ta * dp1dt; o.w = Ta * dp2dt;
    ((float4*)out)[row] = o;
  }
}

extern "C" void kernel_launch(void* const* d_in, const int* in_sizes, int n_in,
                              void* d_out, int out_size, void* d_ws, size_t ws_size,
                              hipStream_t stream) {
  const float* x  = (const float*)d_in[0];
  // d_in[1] = t (unused)
  const float* W1 = (const float*)d_in[2];
  const float* b1 = (const float*)d_in[3];
  const float* W2 = (const float*)d_in[4];
  const float* b2 = (const float*)d_in[5];
  const float* W3 = (const float*)d_in[6];
  const float* b3 = (const float*)d_in[7];
  const float* Tp = (const float*)d_in[8];
  const float* x0 = (const float*)d_in[9];
  float* out = (float*)d_out;
  int N = in_sizes[0] / 4;

  _Float16* w2fT = (_Float16*)d_ws;
  _Float16* w2b  = w2fT + HID * HID;
  float* eDes    = (float*)(w2b + HID * HID);

  dp_prep<<<1, 256, 0, stream>>>(W1, b1, W2, b2, W3, b3, x0, w2fT, w2b, eDes);

  int nTiles = (N + 15) / 16;
  int blocks = (nTiles + 7) / 8;
  dp_main<<<blocks, 256, 0, stream>>>(x, W1, b1, b2, W3, b3, Tp, w2fT, w2b, eDes, out, N);
}
